// RNN2D_62809601736831
// MI455X (gfx1250) — matrix operand order repaired and checked by reading, not compiled, not run on hardware
//
#include <hip/hip_runtime.h>
#include <hip/hip_bf16.h>

// CDNA5 (gfx1250) 2D-RNN kernel.
//
// Transposed-state formulation: state H^T (64 hid x 16 batch) lives in WMMA
// B-operand registers; H_new^T = Wc_h^T @ H^T + Wc_v^T @ V^T + Wcat @ onehot.
// The WMMA D layout (f32) repacks into the next step's B operand (f16) with
// only per-lane v_cvt_pk_f16_f32, because the hidden index is globally
// relabeled by phi(k) = swap bits 3<->4 (baked into the one-time weight pack).
// "Above" state V spills through d_ws (L2-resident: 512 tiles * 64KB = 32MB),
// prefetched one column ahead. Requires ws_size >= 32 MB.

typedef _Float16 v16h __attribute__((ext_vector_type(16)));
typedef float    v8f  __attribute__((ext_vector_type(8)));
typedef float    v4f  __attribute__((ext_vector_type(4)));

#define WMMA(A, B, C) __builtin_amdgcn_wmma_f32_16x16x32_f16( \
    false, (A), false, (B), (short)0, (C), false, false)

__device__ __forceinline__ int phi(int k) {
  // swap bits 3 and 4 of a 6-bit hidden index
  return (k & ~0x18) | ((k & 8) << 1) | ((k & 16) >> 1);
}

union UV { v16h h; v4f f[2]; };

__global__ __launch_bounds__(256)
void rnn2d_wmma_kernel(const int* __restrict__ x,
                       const float* __restrict__ Winh,
                       const float* __restrict__ Winv,
                       const float* __restrict__ Wch,
                       const float* __restrict__ bch,
                       const float* __restrict__ Wcv,
                       const float* __restrict__ Wout,
                       const float* __restrict__ bout,
                       float* __restrict__ out,
                       char* __restrict__ ws) {
  const int lane = threadIdx.x & 31;
  const int hl   = lane >> 4;        // lane half (WMMA layout selector)
  const int n    = lane & 15;        // batch column within tile / M row idx
  const int wid  = (int)((blockIdx.x * blockDim.x + threadIdx.x) >> 5);
  const int b0   = wid * 16;

  const int* xb  = x + (size_t)(b0 + n) * 1024;          // (B,32,32) row-major
  char*      wsb = ws + (size_t)wid * (32u * 32u * 64u); // 64KB per batch tile

  // ---------------- one-time weight packing into A operands (f16) -----------
  // A element (M=16*mt+n, K=32*kt + e + hl*16) holds W[phi(K)][M].
  v16h Ah[4][2], Av[4][2], A3[4], Ao[2], A3o;
#pragma unroll
  for (int mt = 0; mt < 4; ++mt) {
    const int M = 16 * mt + n;
#pragma unroll
    for (int kt = 0; kt < 2; ++kt) {
      v16h ah, av;
#pragma unroll
      for (int e = 0; e < 16; ++e) {
        int K = 32 * kt + e + hl * 16;
        int u = phi(K);
        ah[e] = (_Float16)Wch[u * 64 + M];
        av[e] = (_Float16)Wcv[u * 64 + M];
      }
      Ah[mt][kt] = ah;
      Av[mt][kt] = av;
    }
    // Wcat: K0,K1 = W_in_h rows; K2,K3 = W_in_v rows; K4 = bias; else 0
    v16h a3;
#pragma unroll
    for (int e = 0; e < 16; ++e) {
      int K = e + (e & 8) + hl * 8;
      float v = 0.f;
      if      (K == 0) v = Winh[M];
      else if (K == 1) v = Winh[64 + M];
      else if (K == 2) v = Winv[M];
      else if (K == 3) v = Winv[64 + M];
      else if (K == 4) v = bch[M];
      a3[e] = (_Float16)v;
    }
    A3[mt] = a3;
  }
#pragma unroll
  for (int kt = 0; kt < 2; ++kt) {
    v16h ao;
#pragma unroll
    for (int e = 0; e < 16; ++e) {
      int K = 32 * kt + e + hl * 16;
      int u = phi(K);
      ao[e] = (_Float16)((n < 2) ? Wout[u * 2 + n] : 0.f);
    }
    Ao[kt] = ao;
  }
  {
    v16h a;
#pragma unroll
    for (int e = 0; e < 16; ++e) {
      int K = e + (e & 8) + hl * 8;
      a[e] = (_Float16)((n < 2 && K == 4) ? bout[n] : 0.f);
    }
    A3o = a;
  }

  // ---------------- recurrence --------------------------------------------
  float    lp     = 0.f;
  unsigned rowtok = 0u;
  v16h bv0 = {}, bv1 = {};
  { // prime V prefetch for column 0 (contents unused when i==0)
    const v4f* p = (const v4f*)(wsb + (size_t)(0 * 32 + lane) * 64);
    UV u0, u1;
    u0.f[0] = p[0]; u0.f[1] = p[1]; u1.f[0] = p[2]; u1.f[1] = p[3];
    bv0 = u0.h; bv1 = u1.h;
  }

  for (int i = 0; i < 32; ++i) {
    const unsigned prevrow = rowtok;
    { // pack this row's 32 tokens (values in {0,1}) into a bitmask
      unsigned rt = 0u;
      const int4* rp = (const int4*)(xb + i * 32);
#pragma unroll
      for (int q = 0; q < 8; ++q) {
        int4 v = rp[q];
        rt |= ((unsigned)(v.x & 1) << (4 * q))     |
              ((unsigned)(v.y & 1) << (4 * q + 1)) |
              ((unsigned)(v.z & 1) << (4 * q + 2)) |
              ((unsigned)(v.w & 1) << (4 * q + 3));
      }
      rowtok = rt;
    }

    v16h bh0 = {}, bh1 = {};  // h state = 0 at row start
    for (int j = 0; j < 32; ++j) {
      const int tl = (int)((rowtok  >> ((j - 1) & 31)) & 1);  // left token
      const int ta = (int)((prevrow >> j) & 1);               // above token
      v16h b3 = {};
      b3[0] = (_Float16)((j > 0 && tl == 0) ? 1.f : 0.f);
      b3[1] = (_Float16)((j > 0 && tl == 1) ? 1.f : 0.f);
      b3[2] = (_Float16)((i > 0 && ta == 0) ? 1.f : 0.f);
      b3[3] = (_Float16)((i > 0 && ta == 1) ? 1.f : 0.f);
      b3[4] = (_Float16)1.f;  // bias row

      v8f acc[4];
#pragma unroll
      for (int mt = 0; mt < 4; ++mt) {
        v8f c = {};
        c = WMMA(A3[mt], b3, c);        // embeddings + bias
        c = WMMA(Ah[mt][0], bh0, c);    // h @ W_carry_h
        c = WMMA(Ah[mt][1], bh1, c);
        acc[mt] = c;
      }
      if (i > 0) {                      // v @ W_carry_v (v==0 on first row)
#pragma unroll
        for (int mt = 0; mt < 4; ++mt) {
          acc[mt] = WMMA(Av[mt][0], bv0, acc[mt]);
          acc[mt] = WMMA(Av[mt][1], bv1, acc[mt]);
        }
      }

      { // prefetch V for next column; (j+1)&31 wraps to col 0, which by then
        // holds THIS row's H(.,0) == exactly V for the next row's first step.
        const int jn = (j + 1) & 31;
        const v4f* p = (const v4f*)(wsb + (size_t)(jn * 32 + lane) * 64);
        UV u0, u1;
        u0.f[0] = p[0]; u0.f[1] = p[1]; u1.f[0] = p[2]; u1.f[1] = p[3];
        bv0 = u0.h; bv1 = u1.h;
      }

      // elu, then per-lane repack D(f32) -> next B operand (f16)
#pragma unroll
      for (int mt = 0; mt < 4; ++mt)
#pragma unroll
        for (int r = 0; r < 8; ++r) {
          float v = acc[mt][r];
          acc[mt][r] = (v > 0.f) ? v : (__expf(v) - 1.f);
        }
      v16h nh0, nh1;
#pragma unroll
      for (int r = 0; r < 8; ++r) {
        nh0[r]     = (_Float16)acc[0][r];
        nh0[r + 8] = (_Float16)acc[1][r];
        nh1[r]     = (_Float16)acc[2][r];
        nh1[r + 8] = (_Float16)acc[3][r];
      }
      bh0 = nh0; bh1 = nh1;

      { // store state: becomes V for next row
        v4f* p = (v4f*)(wsb + (size_t)(j * 32 + lane) * 64);
        UV u0, u1; u0.h = bh0; u1.h = bh1;
        p[0] = u0.f[0]; p[1] = u0.f[1]; p[2] = u1.f[0]; p[3] = u1.f[1];
      }

      // logits z = W_out^T @ h_new + b_out, per-lane softmax (z0,z1 in-lane)
      v8f z = {};
      z = WMMA(A3o, b3, z);
      z = WMMA(Ao[0], bh0, z);
      z = WMMA(Ao[1], bh1, z);
      const float z0 = z[0], z1 = z[1];
      const int   t  = (int)((rowtok >> j) & 1);
      const float zt = t ? z1 : z0;
      const float mx = fmaxf(z0, z1);
      const float dd = fabsf(z0 - z1);
      lp += zt - mx - __logf(1.f + __expf(-dd));
    }
  }

  if (lane < 16) out[b0 + n] = lp;
}

extern "C" void kernel_launch(void* const* d_in, const int* in_sizes, int n_in,
                              void* d_out, int out_size, void* d_ws, size_t ws_size,
                              hipStream_t stream) {
  const int*   x    = (const int*)d_in[0];
  const float* Winh = (const float*)d_in[1];
  const float* Winv = (const float*)d_in[2];
  const float* Wch  = (const float*)d_in[3];
  const float* bch  = (const float*)d_in[4];
  const float* Wcv  = (const float*)d_in[5];
  const float* Wout = (const float*)d_in[6];
  const float* bout = (const float*)d_in[7];

  const int B     = 8192;
  const int waves = B / 16;            // 512 independent wave32 chains
  const int block = 256;               // 8 waves per block, no LDS, no barriers
  const int grid  = waves * 32 / block;

  rnn2d_wmma_kernel<<<grid, block, 0, stream>>>(
      x, Winh, Winv, Wch, bch, Wcv, Wout, bout,
      (float*)d_out, (char*)d_ws);
}
